// Head_28819230556856
// MI455X (gfx1250) — matrix operand order repaired and checked by reading, not compiled, not run on hardware
//
#include <hip/hip_runtime.h>

// B=4, T=4096, C=1024, H=64
#define TB 4
#define TT 4096
#define TC 1024
#define TH 64
#define BT (TB*TT)
#define NEGINF (-1e30f)

typedef __attribute__((ext_vector_type(16))) __bf16 v16bf;
typedef __attribute__((ext_vector_type(8)))  float  v8f;

union BF16Frag { v16bf v; unsigned int u[8]; };

// round-half-up fp32 -> bf16 (cheap: add + shift)
__device__ __forceinline__ unsigned short f2bf(float f) {
  return (unsigned short)((__float_as_uint(f) + 0x8000u) >> 16);
}
// pack two fp32 -> bf16x2 with one v_perm_b32
__device__ __forceinline__ unsigned int pk2bf(float lo, float hi) {
  unsigned int ul = __float_as_uint(lo) + 0x8000u;
  unsigned int uh = __float_as_uint(hi) + 0x8000u;
  return __builtin_amdgcn_perm(uh, ul, 0x07060302u);  // {uh[31:16], ul[31:16]}
}

__device__ __forceinline__ float fast_exp2(float x) {
#if __has_builtin(__builtin_amdgcn_exp2f)
  return __builtin_amdgcn_exp2f(x);
#else
  return exp2f(x);
#endif
}

__device__ __forceinline__ v8f wmma_bf16(const BF16Frag& a, const BF16Frag& b, v8f c) {
  return __builtin_amdgcn_wmma_f32_16x16x32_bf16(false, a.v, false, b.v,
                                                 (short)0, c, false, false);
}

// ---------------------------------------------------------------------------
// Kernel 1: W -> bf16, transposed Wt[mat][h][k].
// mat0 = Wq * C^-0.5 * log2(e)  (exp2-rebased logits), 1 = Wk, 2 = Wv.
// ---------------------------------------------------------------------------
__global__ void wt_prep(const float* __restrict__ Wk, const float* __restrict__ Wq,
                        const float* __restrict__ Wv, unsigned short* __restrict__ wt) {
  int idx = blockIdx.x * blockDim.x + threadIdx.x;
  if (idx >= 3 * TC * TH) return;
  int mat = idx / (TC * TH);
  int rem = idx - mat * (TC * TH);
  int h = rem / TC;
  int k = rem - h * TC;
  const float* W = (mat == 0) ? Wq : (mat == 1 ? Wk : Wv);
  float v = W[k * TH + h];
  if (mat == 0) v *= 0.03125f * 1.44269504088896340736f;
  wt[idx] = f2bf(v);
}

// ---------------------------------------------------------------------------
// Kernel 2: QKV projection. One wave per (16-row tile, matrix). 4 n-tiles each.
// Outputs: qb/kb bf16 [BT][64] row-major, vt bf16 [B][64][T] (transposed).
// ---------------------------------------------------------------------------
__global__ void __launch_bounds__(128)
qkv_kernel(const float* __restrict__ x, const unsigned short* __restrict__ wt,
           unsigned short* __restrict__ qb, unsigned short* __restrict__ kb,
           unsigned short* __restrict__ vt) {
  const int lane = threadIdx.x & 31;
  const int wave = threadIdx.x >> 5;
  const int g    = blockIdx.x * 4 + wave;   // 0..3071
  const int mt   = g & 1023;                // 1024 row tiles over BT
  const int mat  = g >> 10;                 // 0=q 1=k 2=v
  const int n0   = lane & 15;
  const int half = lane >> 4;

  const float* xrow = x + (size_t)(mt * 16 + n0) * TC;       // A row per lane
  const unsigned short* wmat = wt + (size_t)mat * TH * TC;

  v8f acc[4];
  #pragma unroll
  for (int t = 0; t < 4; ++t)
    #pragma unroll
    for (int r = 0; r < 8; ++r) acc[t][r] = 0.0f;

  for (int kk = 0; kk < TC; kk += 32) {
    // batch all loads first so the compiler can clause them and use
    // partial s_wait_loadcnt before each WMMA (memory/compute overlap)
    float2 xf[8];
    #pragma unroll
    for (int v = 0; v < 8; ++v) {
      int k = kk + (half ? 16 : 0) + 2 * v;
      xf[v] = *(const float2*)(xrow + k);
    }
    BF16Frag bf[4];
    #pragma unroll
    for (int t = 0; t < 4; ++t) {
      const unsigned short* wcol = wmat + (size_t)(t * 16 + n0) * TC + kk + (half ? 16 : 0);
      #pragma unroll
      for (int v = 0; v < 8; ++v)
        bf[t].u[v] = *(const unsigned int*)(wcol + 2 * v);
    }
    BF16Frag a;
    #pragma unroll
    for (int v = 0; v < 8; ++v) a.u[v] = pk2bf(xf[v].x, xf[v].y);
    #pragma unroll
    for (int t = 0; t < 4; ++t) acc[t] = wmma_bf16(a, bf[t], acc[t]);
  }

  if (mat < 2) {
    unsigned short* dst = (mat == 0) ? qb : kb;
    #pragma unroll
    for (int t = 0; t < 4; ++t)
      #pragma unroll
      for (int r = 0; r < 8; ++r) {
        int m = mt * 16 + r + half * 8;
        dst[(size_t)m * TH + t * 16 + n0] = f2bf(acc[t][r]);
      }
  } else {
    #pragma unroll
    for (int t = 0; t < 4; ++t)
      #pragma unroll
      for (int r = 0; r < 8; ++r) {
        int m = mt * 16 + r + half * 8;         // global token
        int b = m >> 12, tok = m & (TT - 1);
        vt[((size_t)b * TH + t * 16 + n0) * TT + tok] = f2bf(acc[t][r]);
      }
  }
}

// ---------------------------------------------------------------------------
// Kernel 3: causal flash attention. One wave per 16-query tile. kv step = 32.
// Logits arrive pre-scaled by log2(e): softmax uses exp2 directly.
// ---------------------------------------------------------------------------
__global__ void __launch_bounds__(128)
attn_kernel(const unsigned short* __restrict__ qb, const unsigned short* __restrict__ kb,
            const unsigned short* __restrict__ vt, float* __restrict__ out) {
  __shared__ __align__(16) unsigned short plds[4][16 * 32];   // P tile per wave

  const int lane = threadIdx.x & 31;
  const int wave = threadIdx.x >> 5;
  const int tile = blockIdx.x * 4 + wave;   // 0..1023
  const int b    = tile >> 8;
  const int qt   = tile & 255;
  const int n0   = lane & 15;
  const int half = lane >> 4;
  unsigned short* pl = plds[wave];

  // Q A-fragments (h 0..31 and 32..63), held in registers
  BF16Frag aq0, aq1;
  {
    const unsigned short* qrow = qb + (size_t)(b * TT + qt * 16 + n0) * TH;
    #pragma unroll
    for (int v = 0; v < 8; ++v) {
      int h = (half ? 16 : 0) + 2 * v;
      aq0.u[v] = *(const unsigned int*)(qrow + h);
      aq1.u[v] = *(const unsigned int*)(qrow + 32 + h);
    }
  }

  v8f o[4];
  float mrow[8], lrow[8];
  #pragma unroll
  for (int t = 0; t < 4; ++t)
    #pragma unroll
    for (int r = 0; r < 8; ++r) o[t][r] = 0.0f;
  #pragma unroll
  for (int r = 0; r < 8; ++r) { mrow[r] = NEGINF; lrow[r] = 0.0f; }

  auto step = [&](int s0, bool domask) {
    // ---- S = Q K^T for two 16-col tiles: load all 4 K-fragments, then WMMA ----
    BF16Frag bk[4];
    {
      const unsigned short* kr = kb + (size_t)(b * TT + s0 + n0) * TH + (half ? 16 : 0);
      #pragma unroll
      for (int v = 0; v < 8; ++v) bk[0].u[v] = *(const unsigned int*)(kr + 2 * v);
      #pragma unroll
      for (int v = 0; v < 8; ++v) bk[1].u[v] = *(const unsigned int*)(kr + 32 + 2 * v);
      kr += 16 * TH;
      #pragma unroll
      for (int v = 0; v < 8; ++v) bk[2].u[v] = *(const unsigned int*)(kr + 2 * v);
      #pragma unroll
      for (int v = 0; v < 8; ++v) bk[3].u[v] = *(const unsigned int*)(kr + 32 + 2 * v);
    }
    v8f c0, c1;
    #pragma unroll
    for (int r = 0; r < 8; ++r) { c0[r] = 0.0f; c1[r] = 0.0f; }
    c0 = wmma_bf16(aq0, bk[0], c0);
    c0 = wmma_bf16(aq1, bk[1], c0);
    c1 = wmma_bf16(aq0, bk[2], c1);
    c1 = wmma_bf16(aq1, bk[3], c1);

    // ---- causal mask (only the diagonal-crossing step) ----
    if (domask) {
      #pragma unroll
      for (int r = 0; r < 8; ++r) {
        int qr = qt * 16 + r + half * 8;
        if (s0 + n0      > qr) c0[r] = NEGINF;
        if (s0 + 16 + n0 > qr) c1[r] = NEGINF;
      }
    }

    // ---- online softmax (base-2 domain) ----
    #pragma unroll
    for (int r = 0; r < 8; ++r) {
      float rm = fmaxf(c0[r], c1[r]);
      #pragma unroll
      for (int d = 1; d < 16; d <<= 1) rm = fmaxf(rm, __shfl_xor(rm, d, 32));
      float mN = fmaxf(mrow[r], rm);
      float al = fast_exp2(mrow[r] - mN);
      mrow[r] = mN;
      float p0 = fast_exp2(c0[r] - mN);
      float p1 = fast_exp2(c1[r] - mN);
      c0[r] = p0; c1[r] = p1;
      float rs = p0 + p1;
      #pragma unroll
      for (int d = 1; d < 16; d <<= 1) rs += __shfl_xor(rs, d, 32);
      lrow[r] = lrow[r] * al + rs;
      o[0][r] *= al; o[1][r] *= al; o[2][r] *= al; o[3][r] *= al;
    }

    // ---- P (C layout) -> LDS -> A-fragment layout ----
    #pragma unroll
    for (int r = 0; r < 8; ++r) {
      int m = r + half * 8;
      pl[m * 32 + n0]      = f2bf(c0[r]);
      pl[m * 32 + 16 + n0] = f2bf(c1[r]);
    }
    __asm__ volatile("s_wait_dscnt 0" ::: "memory");
    BF16Frag ap;
    #pragma unroll
    for (int v = 0; v < 8; ++v) {
      int s = (half ? 16 : 0) + 2 * v;
      ap.u[v] = *(const unsigned int*)&pl[n0 * 32 + s];
    }

    // ---- O += P * V: load all 4 V-fragments, then WMMA ----
    BF16Frag bv[4];
    #pragma unroll
    for (int t = 0; t < 4; ++t) {
      const unsigned short* vr =
          vt + ((size_t)b * TH + t * 16 + n0) * TT + s0 + (half ? 16 : 0);
      #pragma unroll
      for (int v = 0; v < 8; ++v) bv[t].u[v] = *(const unsigned int*)(vr + 2 * v);
    }
    #pragma unroll
    for (int t = 0; t < 4; ++t) o[t] = wmma_bf16(ap, bv[t], o[t]);
  };

  const int nSteps = ((qt * 16 + 15) >> 5) + 1;
  for (int j = 0; j < nSteps - 1; ++j) step(j * 32, false);   // strictly below diagonal
  step((nSteps - 1) * 32, true);                              // diagonal-crossing step

  // ---- normalize + store fp32 out[B][T][H] ----
  #pragma unroll
  for (int r = 0; r < 8; ++r) {
    float inv = 1.0f / lrow[r];
    int qr = qt * 16 + r + half * 8;
    float* orow = out + ((size_t)(b * TT + qr)) * TH;
    #pragma unroll
    for (int t = 0; t < 4; ++t) orow[t * 16 + n0] = o[t][r] * inv;
  }
}

// ---------------------------------------------------------------------------
extern "C" void kernel_launch(void* const* d_in, const int* in_sizes, int n_in,
                              void* d_out, int out_size, void* d_ws, size_t ws_size,
                              hipStream_t stream) {
  const float* x  = (const float*)d_in[0];
  const float* Wk = (const float*)d_in[1];
  const float* Wq = (const float*)d_in[2];
  const float* Wv = (const float*)d_in[3];

  unsigned short* wt = (unsigned short*)d_ws;       // [3][64][1024]   = 196608
  unsigned short* qb = wt + 3 * TH * TC;            // [BT][64]        = 1048576
  unsigned short* kb = qb + (size_t)BT * TH;        // [BT][64]
  unsigned short* vb = kb + (size_t)BT * TH;        // [B][64][T]
  (void)ws_size; (void)in_sizes; (void)n_in; (void)out_size;

  wt_prep<<<(3 * TC * TH + 255) / 256, 256, 0, stream>>>(Wk, Wq, Wv, wt);
  qkv_kernel<<<(1024 * 3) / 4, 128, 0, stream>>>(x, wt, qb, kb, vb);
  attn_kernel<<<(TB * (TT / 16)) / 4, 128, 0, stream>>>(qb, kb, vb, (float*)d_out);
}
